// MultiHeadAttention_91087666414217
// MI455X (gfx1250) — matrix operand order repaired and checked by reading, not compiled, not run on hardware
//
#include <hip/hip_runtime.h>
#include <hip/hip_bf16.h>

#define NH 8
#define DM 512
#define DK 64
#define B_ 8
#define L_ 2048

typedef __attribute__((ext_vector_type(16))) __bf16 v16bf;
typedef __attribute__((ext_vector_type(8)))  float  v8f;

union FragU { unsigned int u[8]; v16bf v; };
union FB { float f; unsigned int u; };

__device__ __forceinline__ unsigned short f2bf(float x) {
    FB t; t.f = x;
    unsigned int r = (t.u + 0x7FFFu + ((t.u >> 16) & 1u)) >> 16;
    return (unsigned short)r;
}

// A-matrix fragment, 16x32 bf16, row-major source (lda in elements).
// lanes 0-15: M=lane, V0..7 = K 0..15 in pairs
// lanes 16-31: same M, K 16..31
__device__ __forceinline__ v16bf load_a_frag(const unsigned short* p, int lda,
                                             int row0, int k0) {
    int lane = threadIdx.x & 31;
    const unsigned short* r = p + (size_t)(row0 + (lane & 15)) * lda;
    int hk = (lane >> 4) * 16;
    FragU f;
#pragma unroll
    for (int i = 0; i < 4; ++i)
        f.u[i] = *(const unsigned int*)(r + k0 + hk + 2 * i);
#pragma unroll
    for (int i = 0; i < 4; ++i)
        f.u[4 + i] = *(const unsigned int*)(r + k0 + 8 + hk + 2 * i);
    return f.v;
}

// B-matrix fragment, 32x16 bf16, B[k,n] = W[n0+n, k] with W row-major.
// lanes 0-15: N=lane, V0..7 = K 0..15 in pairs; lanes 16-31: K 16..31.
__device__ __forceinline__ v16bf load_b_frag(const unsigned short* p, int lda,
                                             int n0, int k0) {
    int lane = threadIdx.x & 31;
    const unsigned short* r = p + (size_t)(n0 + (lane & 15)) * lda;
    int hk = (lane >> 4) * 16;
    FragU f;
#pragma unroll
    for (int i = 0; i < 8; ++i)
        f.u[i] = *(const unsigned int*)(r + k0 + hk + 2 * i);
    return f.v;
}

__global__ void cvt_bf16_kernel(const float* __restrict__ in,
                                unsigned short* __restrict__ out,
                                float scale, int n) {
    int i = blockIdx.x * 256 + threadIdx.x;
    if (i < n) out[i] = f2bf(in[i] * scale);
}

__global__ void zero_f32_kernel(float* p, int n) {
    int i = blockIdx.x * 256 + threadIdx.x;
    if (i < n) p[i] = 0.0f;
}

// Q = xb @ wqb^T, K = xb @ wkb^T   (M=16384, N=512, K=512), output bf16.
// One wave per 16x64 tile; 2*1024*8 = 16384 wave tasks, 8 waves/block.
__global__ void proj_gemm_kernel(const unsigned short* __restrict__ xb,
                                 const unsigned short* __restrict__ wqb,
                                 const unsigned short* __restrict__ wkb,
                                 unsigned short* __restrict__ Qb,
                                 unsigned short* __restrict__ Kb) {
    int wid = blockIdx.x * 8 + (threadIdx.x >> 5);
    int mat = wid >> 13;       // 0 = Q, 1 = K
    int rem = wid & 8191;
    int m0 = (rem >> 3) * 16;  // 0..16368
    int n0 = (rem & 7) * 64;   // 0..448
    const unsigned short* W = mat ? wkb : wqb;
    unsigned short* Out = mat ? Kb : Qb;

    v8f acc[4];
#pragma unroll
    for (int s = 0; s < 4; ++s) acc[s] = (v8f){0,0,0,0,0,0,0,0};

    for (int kk = 0; kk < DM; kk += 32) {
        v16bf a = load_a_frag(xb, DM, m0, kk);
#pragma unroll
        for (int s = 0; s < 4; ++s) {
            v16bf bf = load_b_frag(W, DM, n0 + s * 16, kk);
            acc[s] = __builtin_amdgcn_wmma_f32_16x16x32_bf16(
                false, a, false, bf, (short)0, acc[s], false, false);
        }
    }
    int lane = threadIdx.x & 31;
    int nl = lane & 15;
    int mh = (lane >> 4) * 8;
#pragma unroll
    for (int s = 0; s < 4; ++s)
#pragma unroll
        for (int r = 0; r < 8; ++r)
            Out[(size_t)(m0 + mh + r) * DM + n0 + s * 16 + nl] = f2bf(acc[s][r]);
}

// pooledRaw[b,h,k] += sum over this block's 16 queries of softmax prob.
// Block = (b, h, qblock); 8 waves x 256 keys each = 2048 keys.
__global__ void attn_pool_kernel(const unsigned short* __restrict__ Qb,
                                 const unsigned short* __restrict__ Kb,
                                 float* __restrict__ pool) {
    __shared__ float smax[8][16];
    __shared__ float ssum[8][16];
    int qb = blockIdx.x & 127;
    int h  = (blockIdx.x >> 7) & 7;
    int b  = blockIdx.x >> 10;
    int wave = threadIdx.x >> 5;
    int lane = threadIdx.x & 31;
    const unsigned short* Qh = Qb + (size_t)b * L_ * DM + h * DK;
    const unsigned short* Kh = Kb + (size_t)b * L_ * DM + h * DK;
    int q0 = qb * 16;
    int k0 = wave * 256;

    v16bf a0 = load_a_frag(Qh, DM, q0, 0);
    v16bf a1 = load_a_frag(Qh, DM, q0, 32);

    v8f acc[16];
#pragma unroll
    for (int t = 0; t < 16; ++t) {
        v8f z = (v8f){0,0,0,0,0,0,0,0};
        v16bf b0 = load_b_frag(Kh, DM, k0 + t * 16, 0);
        z = __builtin_amdgcn_wmma_f32_16x16x32_bf16(false, a0, false, b0,
                                                    (short)0, z, false, false);
        v16bf b1 = load_b_frag(Kh, DM, k0 + t * 16, 32);
        z = __builtin_amdgcn_wmma_f32_16x16x32_bf16(false, a1, false, b1,
                                                    (short)0, z, false, false);
        acc[t] = z;
    }

    int g = lane >> 4;
    // ---- per-row max (rows r + 8g), reduce over 16 tiles then 16 lanes ----
    float vmax[8];
#pragma unroll
    for (int r = 0; r < 8; ++r) {
        float m = acc[0][r];
#pragma unroll
        for (int t = 1; t < 16; ++t) m = fmaxf(m, acc[t][r]);
#pragma unroll
        for (int off = 8; off >= 1; off >>= 1)
            m = fmaxf(m, __shfl_xor(m, off, 32));
        vmax[r] = m;
    }
    if ((lane & 15) == 0) {
#pragma unroll
        for (int r = 0; r < 8; ++r) smax[wave][r + 8 * g] = vmax[r];
    }
    __syncthreads();
    float rm[8];
#pragma unroll
    for (int r = 0; r < 8; ++r) {
        float m = smax[0][r + 8 * g];
        for (int w = 1; w < 8; ++w) m = fmaxf(m, smax[w][r + 8 * g]);
        rm[r] = m;
    }
    // ---- exp + per-row sum ----
    float vs[8];
#pragma unroll
    for (int r = 0; r < 8; ++r) {
        float s = 0.0f;
#pragma unroll
        for (int t = 0; t < 16; ++t) {
            float e = __expf(acc[t][r] - rm[r]);
            acc[t][r] = e;
            s += e;
        }
#pragma unroll
        for (int off = 8; off >= 1; off >>= 1) s += __shfl_xor(s, off, 32);
        vs[r] = s;
    }
    if ((lane & 15) == 0) {
#pragma unroll
        for (int r = 0; r < 8; ++r) ssum[wave][r + 8 * g] = vs[r];
    }
    __syncthreads();
    float ri[8];
#pragma unroll
    for (int r = 0; r < 8; ++r) {
        float s = 0.0f;
        for (int w = 0; w < 8; ++w) s += ssum[w][r + 8 * g];
        ri[r] = 1.0f / s;
    }
    // ---- per-key probability sum over the 16 queries -> pooled ----
    float* ph = pool + ((size_t)(b * NH + h)) * L_ + k0;
#pragma unroll
    for (int t = 0; t < 16; ++t) {
        float p = 0.0f;
#pragma unroll
        for (int r = 0; r < 8; ++r) p += acc[t][r] * ri[r];
        p += __shfl_xor(p, 16, 32);          // combine M=0..7 and M=8..15 halves
        if (lane < 16) atomicAdd(&ph[t * 16 + lane], p);
    }
}

// xp[b,h,:] = (1/L^2) * sum_k pooledRaw[b,h,k] * x[b,k,:]
__global__ void pool_x_kernel(const float* __restrict__ x,
                              const float* __restrict__ pool,
                              float* __restrict__ xp) {
    __shared__ float sp[L_];
    int bh = blockIdx.x;
    int b = bh >> 3;
    const float* pr = pool + (size_t)bh * L_;
    for (int i = threadIdx.x; i < L_; i += 256) sp[i] = pr[i];
    __syncthreads();
    const float* xb = x + (size_t)b * L_ * DM;
    int c0 = threadIdx.x, c1 = threadIdx.x + 256;
    float a0 = 0.0f, a1 = 0.0f;
    for (int k = 0; k < L_; ++k) {
        float p = sp[k];
        a0 += p * xb[(size_t)k * DM + c0];
        a1 += p * xb[(size_t)k * DM + c1];
    }
    const float inv = 1.0f / ((float)L_ * (float)L_);
    xp[(size_t)bh * DM + c0] = a0 * inv;
    xp[(size_t)bh * DM + c1] = a1 * inv;
}

// out chain: w_v contraction, transposed flatten, fc + residual, LayerNorm,
// plus attn_ret. One block (512 threads) per batch element.
__global__ void finalize_kernel(const float* __restrict__ x,
                                const float* __restrict__ w_v,
                                const float* __restrict__ w_fc,
                                const float* __restrict__ gamma,
                                const float* __restrict__ beta,
                                const float* __restrict__ xp,
                                const float* __restrict__ pool,
                                float* __restrict__ out) {
    __shared__ float flat[DM];
    __shared__ float red[DM];
    int b = blockIdx.x, j = threadIdx.x;
    int h = j >> 6, d = j & 63;
    // out[b,h,d] = xp[b,h,:] . w_v[h*64+d,:]  ; flat index = d*NH + h
    const float* xph = xp + (size_t)(b * NH + h) * DM;
    const float* wv = w_v + (size_t)j * DM;
    float o = 0.0f;
    for (int m = 0; m < DM; ++m) o += xph[m] * wv[m];
    flat[d * NH + h] = o;
    __syncthreads();
    // residual = mean over L of x
    const float* xb = x + (size_t)b * L_ * DM;
    float res = 0.0f;
    for (int l = 0; l < L_; ++l) res += xb[(size_t)l * DM + j];
    res *= (1.0f / (float)L_);
    // y = flat @ w_fc^T + residual
    const float* wf = w_fc + (size_t)j * DM;
    float y = res;
    for (int i = 0; i < DM; ++i) y += flat[i] * wf[i];
    // LayerNorm over 512
    red[j] = y;
    __syncthreads();
    for (int s = 256; s > 0; s >>= 1) {
        if (j < s) red[j] += red[j + s];
        __syncthreads();
    }
    float mu = red[0] * (1.0f / (float)DM);
    __syncthreads();
    float dv = y - mu;
    red[j] = dv * dv;
    __syncthreads();
    for (int s = 256; s > 0; s >>= 1) {
        if (j < s) red[j] += red[j + s];
        __syncthreads();
    }
    float var = red[0] * (1.0f / (float)DM);
    out[(size_t)b * DM + j] = dv * rsqrtf(var + 1e-6f) * gamma[j] + beta[j];
    // attn_ret[b,k] = (1/(NH*L)) sum_h pooledRaw[b,h,k]
    float* ar = out + (size_t)B_ * DM + (size_t)b * L_;
    const float* pb = pool + (size_t)b * NH * L_;
    const float inv = 1.0f / ((float)NH * (float)L_);
    for (int k = j; k < L_; k += DM) {
        float s = 0.0f;
        for (int hh = 0; hh < NH; ++hh) s += pb[(size_t)hh * L_ + k];
        ar[k] = s * inv;
    }
}

extern "C" void kernel_launch(void* const* d_in, const int* in_sizes, int n_in,
                              void* d_out, int out_size, void* d_ws, size_t ws_size,
                              hipStream_t stream) {
    const float* x     = (const float*)d_in[0];
    const float* w_q   = (const float*)d_in[1];
    const float* w_k   = (const float*)d_in[2];
    const float* w_v   = (const float*)d_in[3];
    const float* w_fc  = (const float*)d_in[4];
    const float* gamma = (const float*)d_in[5];
    const float* beta  = (const float*)d_in[6];
    float* out = (float*)d_out;
    char* ws = (char*)d_ws;

    const int NX = B_ * L_ * DM;      // 8388608
    const int NW = DM * DM;           // 262144
    const int NPOOL = B_ * NH * L_;   // 131072

    unsigned short* xb  = (unsigned short*)(ws);
    unsigned short* wqb = (unsigned short*)(ws + 16777216);
    unsigned short* wkb = (unsigned short*)(ws + 17301504);
    unsigned short* Qb  = (unsigned short*)(ws + 17825792);
    unsigned short* Kb  = (unsigned short*)(ws + 34603008);
    float* pool = (float*)(ws + 51380224);
    float* xp   = (float*)(ws + 51904512);

    // 1) precision conversion (fold 1/sqrt(DK) into w_q)
    cvt_bf16_kernel<<<NX / 256, 256, 0, stream>>>(x, xb, 1.0f, NX);
    cvt_bf16_kernel<<<NW / 256, 256, 0, stream>>>(w_q, wqb, 0.125f, NW);
    cvt_bf16_kernel<<<NW / 256, 256, 0, stream>>>(w_k, wkb, 1.0f, NW);
    zero_f32_kernel<<<NPOOL / 256, 256, 0, stream>>>(pool, NPOOL);

    // 2) Q/K projections via bf16 WMMA (16384 wave-tiles, 8 waves/block)
    proj_gemm_kernel<<<2048, 256, 0, stream>>>(xb, wqb, wkb, Qb, Kb);

    // 3) attention score WMMA + softmax-pool (B*NH*(L/16) blocks)
    attn_pool_kernel<<<B_ * NH * (L_ / 16), 256, 0, stream>>>(Qb, Kb, pool);

    // 4) pooled-input contraction and finalize
    pool_x_kernel<<<B_ * NH, 256, 0, stream>>>(x, pool, xp);
    finalize_kernel<<<B_, 512, 0, stream>>>(x, w_v, w_fc, gamma, beta, xp, pool, out);
}
